// mfurln_84610855731249
// MI455X (gfx1250) — matrix operand order repaired and checked by reading, not compiled, not run on hardware
//
#include <hip/hip_runtime.h>
#include <hip/hip_bf16.h>

typedef __attribute__((ext_vector_type(16))) _Float16 v16h;
typedef __attribute__((ext_vector_type(8)))  _Float16 v8h;
typedef __attribute__((ext_vector_type(4)))  _Float16 v4h;
typedef __attribute__((ext_vector_type(8)))  float    v8f;
typedef __attribute__((ext_vector_type(4)))  int      v4i;

#if defined(__AMDGCN__) && __has_builtin(__builtin_amdgcn_global_load_async_to_lds_b128) && __has_builtin(__builtin_amdgcn_s_wait_asynccnt)
#define USE_ASYNC_B 1
#else
#define USE_ASYNC_B 0
#endif

#define NB    1024          // batch
#define NPR   28            // pairs
#define NFSZ  2048
#define K1    3072          // NFS+EFS
#define H1D   128
#define K2    3200          // K1 + H1
#define H2D   512
#define MTOT  (NB*NPR)      // 28672

// output offsets (flat, in floats)
#define O_IND 0
#define O_LAB 28672
#define O_LRP 57344
#define O_CRP 98304
#define O_MRP 122880
#define O_LRN 237568
#define O_CRN 339968
#define O_MRN 401408

__constant__ int c_pi[28] = {0,0,0,0,0,0,0,1,1,1,1,1,1,2,2,2,2,2,3,3,3,3,4,4,4,5,5,6};
__constant__ int c_pj[28] = {1,2,3,4,5,6,7,2,3,4,5,6,7,3,4,5,6,7,4,5,6,7,5,6,7,6,7,7};

// ---------- setup: label tensor + row->head-slot map ----------
__global__ void setup_kernel(const int* __restrict__ op, float* __restrict__ label,
                             int* __restrict__ map) {
  int b = blockIdx.x * blockDim.x + threadIdx.x;
  if (b >= NB) return;
  int slot[NPR];
  #pragma unroll
  for (int p = 0; p < NPR; ++p) slot[p] = -1;
  for (int s = 0; s < 8; ++s) {
    int i = op[(b * 8 + s) * 2 + 0];
    int j = op[(b * 8 + s) * 2 + 1];
    int lo = i < j ? i : j;
    int hi = i < j ? j : i;
    int p = lo * (15 - lo) / 2 + (hi - lo - 1);
    slot[p] = s;
  }
  int cnt = 0;
  for (int p = 0; p < NPR; ++p) {
    int m = b * NPR + p;
    if (slot[p] >= 0) { label[m] = 1.0f; map[m] = b * 8 + slot[p]; }
    else              { label[m] = 0.0f; map[m] = NB * 8 + b * 20 + cnt; ++cnt; }
  }
}

// ---------- weight conversion: f32 KxN -> f16 NxK (transposed) ----------
__global__ void convert_kernel(const float* __restrict__ W1, const float* __restrict__ Wr,
                               _Float16* __restrict__ Wt1, _Float16* __restrict__ Wt2) {
  int stride = gridDim.x * blockDim.x;
  int idx = blockIdx.x * blockDim.x + threadIdx.x;
  for (int d = idx; d < K1 * H1D; d += stride) {
    int n = d / K1, k = d - n * K1;
    Wt1[d] = (_Float16)W1[(size_t)k * H1D + n];
  }
  for (int d = idx; d < K2 * H2D; d += stride) {
    int n = d / K2, k = d - n * K2;
    Wt2[d] = (_Float16)Wr[(size_t)k * H2D + n];
  }
}

// ---------- shared helpers ----------
__device__ inline v16h mk_frag(const _Float16* lo, const _Float16* hi) {
  v8h a = *(const v8h*)lo;
  v8h b = *(const v8h*)hi;
  return __builtin_shufflevector(a, b, 0,1,2,3,4,5,6,7,8,9,10,11,12,13,14,15);
}

// ---------- unified region-pipelined GEMM ----------
// Block tile: MROWS x BN, BN = (8/(MROWS/32)) * NT*16.
// Wave grid: (MROWS/32) M-groups x (8/(MROWS/32)) N-groups;
// each wave computes 2 m-tiles x NT n-tiles (NT*16 accum VGPRs).
// A: register prefetch + f32->f16 convert into LDS (double-buffered).
// B: async global->LDS copy (ASYNCcnt) when available, else register staging.
// HAS_IND: tail region [K1,KTOT) read from ind1h (f16).
// !HAS_IND also writes an f16 copy of the output (ind1 path).
template<int MROWS, int NT, int KTOT, bool HAS_IND, int NCOL>
__global__ __launch_bounds__(256) void gemm_kernel(
    const float* __restrict__ nf, const float* __restrict__ inter,
    const _Float16* __restrict__ ind1h,
    const _Float16* __restrict__ Wt, const float* __restrict__ bias,
    float* __restrict__ outf, _Float16* __restrict__ outh) {
  constexpr int WMG = MROWS / 32;        // M wave-groups
  constexpr int WNG = 8 / WMG;           // N wave-groups
  constexpr int BN  = WNG * NT * 16;     // block N width (128 or 256)
  constexpr int ACH = MROWS / 32;        // A chunks per thread
  constexpr int TPR = 256 / BN;          // threads per B row (2 or 1)
  constexpr int PBN = 4 / TPR;           // 16B chunks per thread for B (2 or 4)

  __shared__ _Float16 As[2][MROWS * 40];
  __shared__ _Float16 Bs[2][BN * 40];
  const int tid   = threadIdx.x;
  const int lane  = tid & 31;
  const int wv    = tid >> 5;
  const int wm    = (wv % WMG) * 32;           // wave M offset
  const int wn    = (wv / WMG) * (NT * 16);    // wave N offset
  const int mBase = blockIdx.x * MROWS;
  const int nBase = blockIdx.y * BN;

  v8f acc[2][NT];
  #pragma unroll
  for (int mt = 0; mt < 2; ++mt)
    #pragma unroll
    for (int nt = 0; nt < NT; ++nt)
      #pragma unroll
      for (int r = 0; r < 8; ++r) acc[mt][nt][r] = 0.0f;

  // per-thread A-tile chunk descriptors (ACH chunks of 4 halves each)
  int rowA[ACH], kcA[ACH], bAr[ACH], iAr[ACH], jAr[ACH], mAr[ACH];
  #pragma unroll
  for (int c = 0; c < ACH; ++c) {
    int cid = tid + c * 256;
    rowA[c] = cid >> 3;
    kcA[c]  = (cid & 7) << 2;
    int m = mBase + rowA[c];
    mAr[c] = m;
    int b = m / NPR;
    int p = m - b * NPR;
    bAr[c] = b; iAr[c] = c_pi[p]; jAr[c] = c_pj[p];
  }
  const int bRow = tid / TPR;                 // B row this thread loads
  const int bOff = (tid % TPR) * PBN * 8;     // half-offset within row

  float4 px[ACH], py[ACH];    // A prefetch registers
#if !USE_ASYNC_B
  uint4  pb[PBN];             // B prefetch registers (fallback path)
#endif

  // ---- B tile movement ----
#if USE_ASYNC_B
  auto issueB = [&](int k0, int buf) {   // async global->LDS, no VGPR staging
    const _Float16* g = Wt + (size_t)(nBase + bRow) * KTOT + k0 + bOff;
    _Float16* l = &Bs[buf][bRow * 40 + bOff];
    #pragma unroll
    for (int q = 0; q < PBN; ++q)
      __builtin_amdgcn_global_load_async_to_lds_b128(
          (__attribute__((address_space(1))) v4i*)(g + q * 8),
          (__attribute__((address_space(3))) v4i*)(l + q * 8),
          0, 0);
  };
  auto lateB = [&](int) {};
  auto waitB = [&]() { __builtin_amdgcn_s_wait_asynccnt(0); };
#else
  auto issueB = [&](int k0, int) {
    const uint4* g = (const uint4*)(Wt + (size_t)(nBase + bRow) * KTOT + k0 + bOff);
    #pragma unroll
    for (int q = 0; q < PBN; ++q) pb[q] = g[q];
  };
  auto lateB = [&](int buf) {
    uint4* d = (uint4*)&Bs[buf][bRow * 40 + bOff];
    #pragma unroll
    for (int q = 0; q < PBN; ++q) d[q] = pb[q];
  };
  auto waitB = [&]() {};
#endif

  // ---- per-region A fetch/store (branch-free bodies) ----
  auto fetch_node = [&](int k0) {
    #pragma unroll
    for (int c = 0; c < ACH; ++c) {
      int k = k0 + kcA[c];
      px[c] = *(const float4*)(nf + (((size_t)bAr[c] * 8 + iAr[c]) << 11) + k);
      py[c] = *(const float4*)(nf + (((size_t)bAr[c] * 8 + jAr[c]) << 11) + k);
    }
  };
  auto store_node = [&](int buf) {
    #pragma unroll
    for (int c = 0; c < ACH; ++c) {
      v4h h;
      h[0] = (_Float16)(0.5f * (px[c].x + py[c].x));
      h[1] = (_Float16)(0.5f * (px[c].y + py[c].y));
      h[2] = (_Float16)(0.5f * (px[c].z + py[c].z));
      h[3] = (_Float16)(0.5f * (px[c].w + py[c].w));
      *(v4h*)&As[buf][rowA[c] * 40 + kcA[c]] = h;
    }
  };
  auto fetch_edge = [&](int k0) {
    #pragma unroll
    for (int c = 0; c < ACH; ++c) {
      int k = k0 + kcA[c];
      px[c] = *(const float4*)(inter + (((size_t)bAr[c] * 64 + iAr[c] * 8 + jAr[c]) << 10) + (k - NFSZ));
    }
  };
  auto store_edge = [&](int buf) {
    #pragma unroll
    for (int c = 0; c < ACH; ++c) {
      v4h h;
      h[0] = (_Float16)px[c].x; h[1] = (_Float16)px[c].y;
      h[2] = (_Float16)px[c].z; h[3] = (_Float16)px[c].w;
      *(v4h*)&As[buf][rowA[c] * 40 + kcA[c]] = h;
    }
  };
  auto fetch_ind = [&](int k0) {
    #pragma unroll
    for (int c = 0; c < ACH; ++c) {
      int k = k0 + kcA[c];
      uint2 u = *(const uint2*)(ind1h + (size_t)mAr[c] * H1D + (k - K1));
      px[c].x = __uint_as_float(u.x);
      px[c].y = __uint_as_float(u.y);
    }
  };
  auto store_ind = [&](int buf) {
    #pragma unroll
    for (int c = 0; c < ACH; ++c) {
      float2 t; t.x = px[c].x; t.y = px[c].y;
      *(v4h*)&As[buf][rowA[c] * 40 + kcA[c]] = __builtin_bit_cast(v4h, t);
    }
  };

  auto compute = [&](int buf) {
    const int aLo = (lane >> 4) << 3;
    v16h af[2];
    #pragma unroll
    for (int mt = 0; mt < 2; ++mt) {
      const _Float16* ap = As[buf] + (wm + mt * 16 + (lane & 15)) * 40 + aLo;
      af[mt] = mk_frag(ap, ap + 16);
    }
    const int bLo = (lane >> 4) << 3;
    #pragma unroll
    for (int nt = 0; nt < NT; ++nt) {
      const _Float16* bp = Bs[buf] + (wn + nt * 16 + (lane & 15)) * 40 + bLo;
      v16h bf = mk_frag(bp, bp + 16);
      #pragma unroll
      for (int mt = 0; mt < 2; ++mt)
        acc[mt][nt] = __builtin_amdgcn_wmma_f32_16x16x32_f16(
            false, af[mt], false, bf, (short)0, acc[mt][nt], false, false);
    }
  };

  int buf = 0;
  // ---- region 1: node features [0, NFSZ) ----
  issueB(0, 0); fetch_node(0); store_node(0); lateB(0); waitB();
  for (int k0 = 0; k0 < NFSZ; k0 += 32) {
    __syncthreads();
    int kn = k0 + 32;
    bool more = kn < NFSZ;
    if (more) { issueB(kn, buf ^ 1); fetch_node(kn); }
    compute(buf);
    if (more) { store_node(buf ^ 1); lateB(buf ^ 1); }
    waitB();
    buf ^= 1;
  }
  // ---- region 2: edge features [NFSZ, K1) ----
  issueB(NFSZ, buf); fetch_edge(NFSZ); store_edge(buf); lateB(buf); waitB();
  for (int k0 = NFSZ; k0 < K1; k0 += 32) {
    __syncthreads();
    int kn = k0 + 32;
    bool more = kn < K1;
    if (more) { issueB(kn, buf ^ 1); fetch_edge(kn); }
    compute(buf);
    if (more) { store_edge(buf ^ 1); lateB(buf ^ 1); }
    waitB();
    buf ^= 1;
  }
  // ---- region 3: ind1 tail [K1, KTOT) ----
  if (HAS_IND) {
    issueB(K1, buf); fetch_ind(K1); store_ind(buf); lateB(buf); waitB();
    for (int k0 = K1; k0 < KTOT; k0 += 32) {
      __syncthreads();
      int kn = k0 + 32;
      bool more = kn < KTOT;
      if (more) { issueB(kn, buf ^ 1); fetch_ind(kn); }
      compute(buf);
      if (more) { store_ind(buf ^ 1); lateB(buf ^ 1); }
      waitB();
      buf ^= 1;
    }
  }

  // epilogue
  const int rHi = (lane >> 4) << 3;
  const int nc  = lane & 15;
  #pragma unroll
  for (int mt = 0; mt < 2; ++mt)
    #pragma unroll
    for (int nt = 0; nt < NT; ++nt) {
      int n = nBase + wn + nt * 16 + nc;
      float bv = bias[n];
      #pragma unroll
      for (int r = 0; r < 8; ++r) {
        int m = mBase + wm + mt * 16 + rHi + r;
        float val = acc[mt][nt][r] + bv;
        outf[(size_t)m * NCOL + n] = val;
        if (!HAS_IND) outh[(size_t)m * NCOL + n] = (_Float16)val;
      }
    }
}

// ---------- indeterminate head: sigmoid(ind1 @ W_ind2 + b) ----------
__global__ __launch_bounds__(256) void indet_kernel(const float* __restrict__ ind1,
                                                    const float* __restrict__ W2,
                                                    const float* __restrict__ b2,
                                                    float* __restrict__ out) {
  int wv = threadIdx.x >> 5, lane = threadIdx.x & 31;
  int m = blockIdx.x * 8 + wv;
  float4 x = ((const float4*)(ind1 + (size_t)m * H1D))[lane];
  float4 w = ((const float4*)W2)[lane];
  float s = x.x * w.x + x.y * w.y + x.z * w.z + x.w * w.w;
  #pragma unroll
  for (int off = 16; off > 0; off >>= 1) s += __shfl_xor(s, off, 32);
  if (lane == 0) out[m] = 1.0f / (1.0f + __expf(-(s + b2[0])));
}

// ---------- heads: rel row (512) -> lr(5)/cr(3)/mr(14), scattered pos/neg ----------
__global__ __launch_bounds__(256) void head_kernel(
    const float* __restrict__ rel, const int* __restrict__ map,
    const float* __restrict__ Wlr, const float* __restrict__ blr,
    const float* __restrict__ Wcr, const float* __restrict__ bcr,
    const float* __restrict__ Wmr, const float* __restrict__ bmr,
    float* __restrict__ out) {
  __shared__ float rs[8 * H2D];
  int t = threadIdx.x;
  int wv = t >> 5, lane = t & 31;
  int m0 = blockIdx.x * 8;
  const float4* src = (const float4*)(rel + (size_t)(m0 + wv) * H2D);
  float4* dst = (float4*)(rs + wv * H2D);
  #pragma unroll
  for (int c = 0; c < 4; ++c) dst[lane + c * 32] = src[lane + c * 32];
  __syncthreads();
  if (t < 8 * 22) {
    int r = t / 22, c = t - r * 22;
    int m = m0 + r;
    const float* row = rs + r * H2D;
    int rid = map[m];
    bool pos = rid < NB * 8;
    int nid = rid - NB * 8;
    const float* W; const float* bias; int NO, o; size_t obase;
    if (c < 5)      { W = Wlr; bias = blr; NO = 5;  o = c;
                      obase = pos ? (O_LRP + (size_t)rid * 5)  : (O_LRN + (size_t)nid * 5); }
    else if (c < 8) { W = Wcr; bias = bcr; NO = 3;  o = c - 5;
                      obase = pos ? (O_CRP + (size_t)rid * 3)  : (O_CRN + (size_t)nid * 3); }
    else            { W = Wmr; bias = bmr; NO = 14; o = c - 8;
                      obase = pos ? (O_MRP + (size_t)rid * 14) : (O_MRN + (size_t)nid * 14); }
    float s = bias[o];
    for (int k = 0; k < H2D; ++k) s += row[k] * W[k * NO + o];
    out[obase + o] = s;
  }
}

extern "C" void kernel_launch(void* const* d_in, const int* in_sizes, int n_in,
                              void* d_out, int out_size, void* d_ws, size_t ws_size,
                              hipStream_t stream) {
  (void)in_sizes; (void)n_in; (void)out_size; (void)ws_size;
  const float* nf    = (const float*)d_in[0];
  const float* inter = (const float*)d_in[1];
  const int*   op    = (const int*)d_in[2];
  const float* W1    = (const float*)d_in[3];
  const float* b1    = (const float*)d_in[4];
  const float* W2    = (const float*)d_in[5];
  const float* b2    = (const float*)d_in[6];
  const float* Wr    = (const float*)d_in[7];
  const float* br    = (const float*)d_in[8];
  const float* Wcr   = (const float*)d_in[9];
  const float* bcr   = (const float*)d_in[10];
  const float* Wlr   = (const float*)d_in[11];
  const float* blr   = (const float*)d_in[12];
  const float* Wmr   = (const float*)d_in[13];
  const float* bmr   = (const float*)d_in[14];
  float* out = (float*)d_out;
  char* ws = (char*)d_ws;

  // workspace layout (all offsets multiples of 256B), total ~81 MB
  _Float16* Wt1   = (_Float16*)(ws + 0);          //  786,432 B
  _Float16* Wt2   = (_Float16*)(ws + 786432);     //  3,276,800 B
  float*    ind1  = (float*)   (ws + 4063232);    // 14,680,064 B
  _Float16* ind1h = (_Float16*)(ws + 18743296);   //  7,340,032 B
  float*    rel   = (float*)   (ws + 26083328);   // 58,720,256 B
  int*      map   = (int*)     (ws + 84803584);   //    114,688 B

  setup_kernel<<<4, 256, 0, stream>>>(op, out + O_LAB, map);
  convert_kernel<<<2048, 256, 0, stream>>>(W1, Wr, Wt1, Wt2);
  // GEMM1: 64x128 block tile (NT=2), K=3072
  gemm_kernel<64, 2, K1, false, H1D><<<MTOT / 64, 256, 0, stream>>>(
      nf, inter, nullptr, Wt1, b1, ind1, ind1h);
  indet_kernel<<<MTOT / 8, 256, 0, stream>>>(ind1, W2, b2, out + O_IND);
  // GEMM2: 64x256 block tile (NT=4), K=3200 (feature + ind1 tail)
  gemm_kernel<64, 4, K2, true, H2D><<<dim3(MTOT / 64, 2), 256, 0, stream>>>(
      nf, inter, ind1h, Wt2, br, rel, nullptr);
  head_kernel<<<MTOT / 8, 256, 0, stream>>>(rel, map, Wlr, blr, Wcr, bcr, Wmr, bmr, out);
}
